// SelfAttention_67147518706060
// MI455X (gfx1250) — matrix operand order repaired and checked by reading, not compiled, not run on hardware
//
#include <hip/hip_runtime.h>
#include <hip/hip_bf16.h>

// ---------------------------------------------------------------------------
// Problem constants (from reference): B=4, H=W=64 -> N=4096, C=256, DK=32
// ---------------------------------------------------------------------------
#define BB 4
#define NN 4096
#define CC 256
#define DKK 32
#define WAVES 8                   // 8 waves * 16 q-rows = 128 q rows / block

typedef __attribute__((ext_vector_type(8)))  float        v8f;
typedef __attribute__((ext_vector_type(8)))  unsigned int v8u;
typedef __attribute__((ext_vector_type(16))) __bf16       v16bf;

__device__ __forceinline__ unsigned short f32_to_bf16(float f) {
    unsigned int u = __float_as_uint(f);
    unsigned int r = (u + 0x7FFFu + ((u >> 16) & 1u)) >> 16;
    return (unsigned short)r;
}

// Pack two f32 -> two bf16 in one dword: round-bias adds + one v_perm_b32.
__device__ __forceinline__ unsigned int pack_bf16x2(float lo, float hi) {
    unsigned int a = __float_as_uint(lo) + 0x8000u;
    unsigned int b = __float_as_uint(hi) + 0x8000u;
    return __builtin_amdgcn_perm(b, a, 0x07060302u);
}

// Load 16 bf16 (two 16-byte chunks) into a v16bf operand register set.
// Works for both global and LDS source pointers (addrspace inferred).
__device__ __forceinline__ v16bf load_bf16x16(const unsigned short* p0,
                                              const unsigned short* p1) {
    uint4 a = *(const uint4*)p0;
    uint4 b = *(const uint4*)p1;
    v8u u;
    u[0] = a.x; u[1] = a.y; u[2] = a.z; u[3] = a.w;
    u[4] = b.x; u[5] = b.y; u[6] = b.z; u[7] = b.w;
    return __builtin_bit_cast(v16bf, u);
}

// 16-lane-half butterfly reductions (used ONCE per kernel, off the hot loop).
__device__ __forceinline__ float rowmax16(float v) {
    v = fmaxf(v, __shfl_xor(v, 1, 32));
    v = fmaxf(v, __shfl_xor(v, 2, 32));
    v = fmaxf(v, __shfl_xor(v, 4, 32));
    v = fmaxf(v, __shfl_xor(v, 8, 32));
    return v;
}
__device__ __forceinline__ float rowsum16(float v) {
    v += __shfl_xor(v, 1, 32);
    v += __shfl_xor(v, 2, 32);
    v += __shfl_xor(v, 4, 32);
    v += __shfl_xor(v, 8, 32);
    return v;
}

// ---- CDNA5 async global->LDS copies (ASYNCcnt-tracked) --------------------
// LDS destination address = low 32 bits of the generic pointer to __shared__.
__device__ __forceinline__ void async_ld_b128(unsigned lds_off, const void* g) {
    asm volatile("global_load_async_to_lds_b128 %0, %1, off"
                 :: "v"(lds_off), "v"((unsigned long long)(size_t)g) : "memory");
}
__device__ __forceinline__ void async_ld_b64(unsigned lds_off, const void* g) {
    asm volatile("global_load_async_to_lds_b64 %0, %1, off"
                 :: "v"(lds_off), "v"((unsigned long long)(size_t)g) : "memory");
}
__device__ __forceinline__ unsigned lds_off_of(const void* p) {
    return (unsigned)(size_t)p;
}

// ---------------------------------------------------------------------------
// Phase 1: per-pixel projections. q = (x@Wq+bq) * log2(e)/sqrt(dk),
// k = x@Wk+bk, v = x@Wv+bv.  Emits bf16: Q[B][N][DK], K[B][N][DK] and
// V TRANSPOSED Vt[B][C][N] so phase-2 B-operand rows are contiguous.
// One block per (b, n) pixel; 320 threads = 32(q) + 32(k) + 256(v).
// ---------------------------------------------------------------------------
__global__ __launch_bounds__(320) void proj_qkv_kernel(
    const float* __restrict__ x,
    const float* __restrict__ Wq, const float* __restrict__ bq,
    const float* __restrict__ Wk, const float* __restrict__ bk,
    const float* __restrict__ Wv, const float* __restrict__ bv,
    unsigned short* __restrict__ Qg,
    unsigned short* __restrict__ Kg,
    unsigned short* __restrict__ Vtg) {
    __shared__ float xs[CC];
    const int bn = blockIdx.x;          // 0 .. B*N-1
    const int b  = bn / NN;
    const int n  = bn % NN;
    const float* xr = x + (size_t)bn * CC;
    for (int i = threadIdx.x; i < CC; i += blockDim.x) xs[i] = xr[i];
    __syncthreads();

    const int j = threadIdx.x;
    if (j < DKK) {                       // Q column j  (fold 1/sqrt(dk)*log2e)
        float a = bq[j];
#pragma unroll 8
        for (int c = 0; c < CC; ++c) a = fmaf(xs[c], Wq[c * DKK + j], a);
        Qg[(size_t)bn * DKK + j] =
            f32_to_bf16(a * (0.17677669529663687f * 1.4426950408889634f));
    } else if (j < 2 * DKK) {            // K column j-32
        const int jj = j - DKK;
        float a = bk[jj];
#pragma unroll 8
        for (int c = 0; c < CC; ++c) a = fmaf(xs[c], Wk[c * DKK + jj], a);
        Kg[(size_t)bn * DKK + jj] = f32_to_bf16(a);
    } else {                             // V column j-64, stored transposed
        const int jj = j - 2 * DKK;
        float a = bv[jj];
#pragma unroll 8
        for (int c = 0; c < CC; ++c) a = fmaf(xs[c], Wv[c * CC + jj], a);
        Vtg[((size_t)b * CC + jj) * NN + n] = f32_to_bf16(a);
    }
}

// ---------------------------------------------------------------------------
// Phase 2: two-pass (offline-max) flash attention with async-LDS staging.
// Workgroup = 8 waves x 16 q-rows = 128 rows.  Per 32-key block the
// workgroup cooperatively async-copies the K tile (32x32 bf16, 2KB) and the
// V^T tile (256x32 bf16, 16KB) into double-buffered LDS (one copy instead of
// eight per-wave GL2 reads), then every wave consumes B-operands from LDS.
//   Pass 1: 2x wmma_bf16 per block, per-lane running max only.
//   Pass 2: 18x wmma_bf16 per block, no rescale, per-lane l accumulation.
// Cross-lane reductions happen exactly once (after each pass).
// Epilogue: out = gamma * (acc / l) + x.
// ---------------------------------------------------------------------------
__global__ __launch_bounds__(32 * WAVES) void flash_attn_kernel(
    const unsigned short* __restrict__ Qg,
    const unsigned short* __restrict__ Kg,
    const unsigned short* __restrict__ Vtg,
    const float* __restrict__ x,
    const float* __restrict__ gamma,
    float* __restrict__ out) {
    const int wave = threadIdx.x >> 5;
    const int lane = threadIdx.x & 31;
    const int half = lane >> 4;          // 0: lanes 0-15, 1: lanes 16-31
    const int ln   = lane & 15;

    const int blocksPerBatch = NN / (16 * WAVES);
    const int b     = blockIdx.x / blocksPerBatch;
    const int qbase = (blockIdx.x % blocksPerBatch) * (16 * WAVES) + wave * 16;

    // Double-buffered cooperative tiles + per-wave P staging.
    __shared__ unsigned short Klds[2][32][32];    //  4 KB: 32 keys x 64B
    __shared__ unsigned short Vlds[2][CC][32];    // 32 KB: 256 ch  x 64B
    __shared__ unsigned int   Plds[WAVES][16][16];//  8 KB

    const unsigned kbuf0 = lds_off_of(&Klds[0][0][0]);
    const unsigned kbuf1 = lds_off_of(&Klds[1][0][0]);
    const unsigned vbuf0 = lds_off_of(&Vlds[0][0][0]);
    const unsigned vbuf1 = lds_off_of(&Vlds[1][0][0]);

    // ---- Q A-operand (16x32 bf16): lane half selects K sub-bands ----------
    const unsigned short* qrow = Qg + ((size_t)b * NN + qbase + ln) * DKK;
    const v16bf Aq = load_bf16x16(qrow + half * 16, qrow + half * 16 + 8);

    const unsigned short* kbatch = Kg  + (size_t)b * NN * DKK;
    const unsigned short* vbatch = Vtg + (size_t)b * CC * NN;

    // Async-issue slices (uniform per wave: 1x b64 K + 4x b128 V = 5 ops).
    const int kkey = wave * 4 + (lane >> 3);   // this lane's K row (of 32)
    const int ks8  = lane & 7;                 // 8B segment in 64B row
    const int vseg = lane & 3;                 // 16B segment in 64B row
    const int vcr  = lane >> 2;                // 0..7 channel-in-group

#define ISSUE_K(buf, kb)                                                     \
    async_ld_b64((buf) + kkey * 64 + ks8 * 8,                                \
                 kbatch + (size_t)((kb) + kkey) * DKK + ks8 * 4)
#define ISSUE_V(buf, kb)                                                     \
    _Pragma("unroll")                                                        \
    for (int j = 0; j < 4; ++j) {                                            \
        const int ch = wave * 32 + j * 8 + vcr;                              \
        async_ld_b128((buf) + ch * 64 + vseg * 16,                           \
                      vbatch + (size_t)ch * NN + (kb) + vseg * 8);           \
    }

    // =================== pass 1: row max (per-lane, reduce once) ==========
    float m[8];
#pragma unroll
    for (int r = 0; r < 8; ++r) m[r] = -INFINITY;

    ISSUE_K(kbuf0, 0);
    int cur = 0;
#pragma unroll 1
    for (int kb = 0; kb < NN; kb += 32) {
        if (kb + 32 < NN) {
            ISSUE_K(cur ? kbuf0 : kbuf1, kb + 32);
            asm volatile("s_wait_asynccnt 0x1" ::: "memory");
        } else {
            asm volatile("s_wait_asynccnt 0x0" ::: "memory");
        }
        __syncthreads();

        const v16bf Bk0 = load_bf16x16(&Klds[cur][2 * ln][half * 16],
                                       &Klds[cur][2 * ln][half * 16] + 8);
        const v16bf Bk1 = load_bf16x16(&Klds[cur][2 * ln + 1][half * 16],
                                       &Klds[cur][2 * ln + 1][half * 16] + 8);
        v8f z = (v8f){};
        v8f S0 = __builtin_amdgcn_wmma_f32_16x16x32_bf16(false, Aq, false, Bk0,
                                                         (short)0, z, false, false);
        v8f S1 = __builtin_amdgcn_wmma_f32_16x16x32_bf16(false, Aq, false, Bk1,
                                                         (short)0, z, false, false);
#pragma unroll
        for (int r = 0; r < 8; ++r)
            m[r] = fmaxf(m[r], fmaxf(S0[r], S1[r]));

        __syncthreads();          // buffer safe to overwrite next iteration
        cur ^= 1;
    }
#pragma unroll
    for (int r = 0; r < 8; ++r) m[r] = rowmax16(m[r]);   // exact row max

    // =================== pass 2: accumulate O and l (no rescale) ==========
    float l[8];
    v8f acc[16];
#pragma unroll
    for (int r = 0; r < 8; ++r) l[r] = 0.0f;
#pragma unroll
    for (int t = 0; t < 16; ++t) acc[t] = (v8f){};

    ISSUE_K(kbuf0, 0);
    ISSUE_V(vbuf0, 0);
    cur = 0;
#pragma unroll 1
    for (int kb = 0; kb < NN; kb += 32) {
        if (kb + 32 < NN) {
            ISSUE_K(cur ? kbuf0 : kbuf1, kb + 32);
            ISSUE_V(cur ? vbuf0 : vbuf1, kb + 32);
            asm volatile("s_wait_asynccnt 0x5" ::: "memory");
        } else {
            asm volatile("s_wait_asynccnt 0x0" ::: "memory");
        }
        __syncthreads();

        // ---- S = Q K^T (even/odd key tiles) from LDS ---------------------
        const v16bf Bk0 = load_bf16x16(&Klds[cur][2 * ln][half * 16],
                                       &Klds[cur][2 * ln][half * 16] + 8);
        const v16bf Bk1 = load_bf16x16(&Klds[cur][2 * ln + 1][half * 16],
                                       &Klds[cur][2 * ln + 1][half * 16] + 8);
        v8f z = (v8f){};
        v8f S0 = __builtin_amdgcn_wmma_f32_16x16x32_bf16(false, Aq, false, Bk0,
                                                         (short)0, z, false, false);
        v8f S1 = __builtin_amdgcn_wmma_f32_16x16x32_bf16(false, Aq, false, Bk1,
                                                         (short)0, z, false, false);

        // P = 2^(S - m): v_exp_f32 is natively base-2 (log2e folded into Q).
#pragma unroll
        for (int r = 0; r < 8; ++r) {
            const float p0 = exp2f(S0[r] - m[r]);   // even key
            const float p1 = exp2f(S1[r] - m[r]);   // odd key
            l[r] += p0 + p1;
            Plds[wave][r + half * 8][ln] = pack_bf16x2(p0, p1);
        }

        // C-layout -> A-layout via LDS; fence this wave's DS stores.
        asm volatile("s_wait_dscnt 0x0" ::: "memory");
        const unsigned short* prow = (const unsigned short*)&Plds[wave][ln][0];
        const v16bf Ap = load_bf16x16(prow + half * 16, prow + half * 16 + 8);

        // ---- O += P * V : 16 channel tiles, B-operands from LDS ----------
#pragma unroll
        for (int t = 0; t < 16; ++t) {
            const v16bf Bv = load_bf16x16(&Vlds[cur][t * 16 + ln][half * 16],
                                          &Vlds[cur][t * 16 + ln][half * 16] + 8);
            acc[t] = __builtin_amdgcn_wmma_f32_16x16x32_bf16(false, Ap, false, Bv,
                                                             (short)0, acc[t],
                                                             false, false);
        }

        __syncthreads();          // buffers safe to overwrite next iteration
        cur ^= 1;
    }

    // ---- epilogue: reduce l once, out = gamma * (acc / l) + x ------------
    const float g = gamma[0];
#pragma unroll
    for (int r = 0; r < 8; ++r) {
        const float inv = 1.0f / rowsum16(l[r]);
        const int row   = qbase + r + half * 8;
        const size_t base = ((size_t)b * NN + row) * CC;
#pragma unroll
        for (int t = 0; t < 16; ++t) {
            const int ch = t * 16 + ln;
            out[base + ch] = fmaf(g, acc[t][r] * inv, x[base + ch]);
        }
    }
#undef ISSUE_K
#undef ISSUE_V
}

// ---------------------------------------------------------------------------
extern "C" void kernel_launch(void* const* d_in, const int* in_sizes, int n_in,
                              void* d_out, int out_size, void* d_ws, size_t ws_size,
                              hipStream_t stream) {
    const float* x     = (const float*)d_in[0];
    const float* Wq    = (const float*)d_in[1];
    const float* bq    = (const float*)d_in[2];
    const float* Wk    = (const float*)d_in[3];
    const float* bk    = (const float*)d_in[4];
    const float* Wv    = (const float*)d_in[5];
    const float* bv    = (const float*)d_in[6];
    const float* gamma = (const float*)d_in[7];
    float* out = (float*)d_out;

    // Workspace layout (bf16): Q[B][N][DK] | K[B][N][DK] | Vt[B][C][N]  (~10 MB)
    unsigned short* Qg  = (unsigned short*)d_ws;
    unsigned short* Kg  = Qg + (size_t)BB * NN * DKK;
    unsigned short* Vtg = Kg + (size_t)BB * NN * DKK;

    proj_qkv_kernel<<<BB * NN, 320, 0, stream>>>(x, Wq, bq, Wk, bk, Wv, bv,
                                                 Qg, Kg, Vtg);

    const int grid = (BB * NN) / (16 * WAVES);   // 128 workgroups, exact
    flash_attn_kernel<<<grid, 32 * WAVES, 0, stream>>>(Qg, Kg, Vtg, x, gamma, out);
}
